// EncoderCoreDecoder_18751827214708
// MI455X (gfx1250) — compile-verified, matrix operand order checked by reading, not run on hardware
//
#include <hip/hip_runtime.h>
#include <cstdint>
#include <cstddef>

// ---------------------------------------------------------------------------
// GraphNet Encode-Process-Decode for MI455X (gfx1250), wave32 + WMMA f16.
// ---------------------------------------------------------------------------

typedef __attribute__((ext_vector_type(16))) _Float16 v16h;
typedef __attribute__((ext_vector_type(8)))  _Float16 v8h;
typedef __attribute__((ext_vector_type(8)))  float    v8f;

static constexpr int kNV = 50000;
static constexpr int kNE = 400000;
static constexpr int kNU = 128;

// ------------------------------- device helpers ----------------------------

__device__ __forceinline__ v8f wmma16x16x32(const v16h& a, const v16h& b, const v8f& c) {
  // D = A(16x32 f16) x B(32x16 f16) + C(16x16 f32)
  return __builtin_amdgcn_wmma_f32_16x16x32_f16(false, a, false, b, (short)0, c, false, false);
}

__device__ __forceinline__ v8f zero8() {
  v8f z;
#pragma unroll
  for (int i = 0; i < 8; ++i) z[i] = 0.f;
  return z;
}

__device__ __forceinline__ v8h load8h(const _Float16* p) { return *(const v8h*)p; }

__device__ __forceinline__ v8h cvt8(const float* p) {
  const float4* q = (const float4*)p;
  float4 a = q[0], b = q[1];
  v8h r;
  r[0] = (_Float16)a.x; r[1] = (_Float16)a.y; r[2] = (_Float16)a.z; r[3] = (_Float16)a.w;
  r[4] = (_Float16)b.x; r[5] = (_Float16)b.y; r[6] = (_Float16)b.z; r[7] = (_Float16)b.w;
  return r;
}

__device__ __forceinline__ v16h mk16(v8h lo, v8h hi) {
  v16h r;
#pragma unroll
  for (int i = 0; i < 8; ++i) { r[i] = lo[i]; r[i + 8] = hi[i]; }
  return r;
}

// ------------------------------- MLP kernel --------------------------------

struct MlpArgs {
  const float*    src;      // MODE 0: row-major [M x K1] f32
  const _Float16* xv64;     // [NV x 64] f16 (enc_x | lat_v)
  const _Float16* ea64;     // [NE x 64] f16
  const _Float16* uu64;     // [NU x 64] f16
  const int*      row;      // edge src
  const int*      col;      // edge dst
  const int*      eidx;     // e_indices
  const int*      vidx;     // v_indices
  const float*    agg;      // [NV x 32] f32
  const float*    vsum;     // [NU x 32]
  const float*    vcnt;     // [NU]
  const float*    esum;     // [NU x 32]
  const float*    ecnt;     // [NU]
  const _Float16* W1t;      // [64 x K1] f16 (transposed)
  const float*    b1;       // [64]
  const _Float16* W2t;      // [32 x 64] f16
  const float*    b2;       // [32]
  const float*    g;        // [32]
  const float*    be;       // [32]
  const _Float16* W3t;      // [32 x 32] f16 (fused output linear)
  const float*    b3;       // [32]
  float*          out;      // [M x 32]
  int             M;
};

// A-fragment source: 8 contiguous (logical) halves at (row m, col k).
template <int K1, int MODE>
__device__ __forceinline__ v8h fetch8(const MlpArgs& A, int m, int k) {
  if constexpr (MODE == 0) {                      // plain f32 matrix
    return cvt8(A.src + (size_t)m * K1 + k);
  } else if constexpr (MODE == 1) {               // edge cat: xv[row]|xv[col]|ea|uu[eidx]
    int q = k >> 6, kk = k & 63;
    const _Float16* p;
    if (q == 0)      p = A.xv64 + (size_t)A.row[m] * 64 + kk;
    else if (q == 1) p = A.xv64 + (size_t)A.col[m] * 64 + kk;
    else if (q == 2) p = A.ea64 + (size_t)m * 64 + kk;
    else             p = A.uu64 + (size_t)A.eidx[m] * 64 + kk;
    return load8h(p);
  } else if constexpr (MODE == 2) {               // node cat: xv|agg|uu[vidx]
    if (k < 64)      return load8h(A.xv64 + (size_t)m * 64 + k);
    else if (k < 96) return cvt8(A.agg + (size_t)m * 32 + (k - 64));
    else             return load8h(A.uu64 + (size_t)A.vidx[m] * 64 + (k - 96));
  } else {                                        // glob cat: uu|vmean|emean
    if (k < 64) return load8h(A.uu64 + (size_t)m * 64 + k);
    if (k < 96) {
      float rc = 1.f / fmaxf(A.vcnt[m], 1.f);
      const float* s = A.vsum + (size_t)m * 32 + (k - 64);
      v8h r;
#pragma unroll
      for (int i = 0; i < 8; ++i) r[i] = (_Float16)(s[i] * rc);
      return r;
    }
    float rc = 1.f / fmaxf(A.ecnt[m], 1.f);
    const float* s = A.esum + (size_t)m * 32 + (k - 96);
    v8h r;
#pragma unroll
    for (int i = 0; i < 8; ++i) r[i] = (_Float16)(s[i] * rc);
    return r;
  }
}

// Two-layer MLP (K1->64 ReLU, 64->32 ReLU) + LayerNorm(32) [+ fused 32->32 linear].
// One wave computes 16 rows; 8 waves per 256-thread block.
template <int K1, int MODE, bool FUSE3>
__global__ __launch_bounds__(256) void mlp_kernel(MlpArgs A) {
  __shared__ __align__(16) _Float16 sW1[64 * K1];
  __shared__ __align__(16) _Float16 sW2[32 * 64];
  __shared__ __align__(16) _Float16 sW3[FUSE3 ? 32 * 32 : 8];
  __shared__ __align__(16) _Float16 sH[8 * 16 * 64];

  const int tid = threadIdx.x;

  // cooperative stage of weights into LDS (16B chunks)
  {
    const int4* g1 = (const int4*)A.W1t;
    int4* s1 = (int4*)sW1;
    for (int i = tid; i < 64 * K1 / 8; i += 256) s1[i] = g1[i];
    const int4* g2 = (const int4*)A.W2t;
    int4* s2 = (int4*)sW2;
    for (int i = tid; i < 32 * 64 / 8; i += 256) s2[i] = g2[i];
    if constexpr (FUSE3) {
      const int4* g3 = (const int4*)A.W3t;
      int4* s3 = (int4*)sW3;
      for (int i = tid; i < 32 * 32 / 8; i += 256) s3[i] = g3[i];
    }
  }
  __syncthreads();

  const int wave = tid >> 5, lane = tid & 31;
  const int lm = lane & 15, hi = lane >> 4;        // hi selects K-subgroup
  const int m0 = (blockIdx.x * 8 + wave) * 16;
  const bool active = (m0 < A.M);                  // all M are multiples of 16
  _Float16* hW = sH + wave * 16 * 64;

  if (active) {
    // ---- layer 1: [16 x K1] x [K1 x 64] ----
    v8f acc[4];
#pragma unroll
    for (int nt = 0; nt < 4; ++nt) acc[nt] = zero8();
    const int m = m0 + lm;
#pragma unroll
    for (int kt = 0; kt < K1 / 32; ++kt) {
      const int k0 = kt * 32, kb = k0 + hi * 8;
      v16h a = mk16(fetch8<K1, MODE>(A, m, kb), fetch8<K1, MODE>(A, m, kb + 16));
#pragma unroll
      for (int nt = 0; nt < 4; ++nt) {
        const _Float16* wr = sW1 + (size_t)(nt * 16 + lm) * K1 + kb;
        v16h b = mk16(load8h(wr), load8h(wr + 16));
        acc[nt] = wmma16x16x32(a, b, acc[nt]);
      }
    }
    // bias + ReLU -> f16 tile (row-major 16x64) in LDS
#pragma unroll
    for (int nt = 0; nt < 4; ++nt) {
      const int c = nt * 16 + lm;
      const float bb = A.b1[c];
#pragma unroll
      for (int r = 0; r < 8; ++r) {
        float v = fmaxf(acc[nt][r] + bb, 0.f);
        hW[(size_t)(hi * 8 + r) * 64 + c] = (_Float16)v;
      }
    }
  }
  __syncthreads();

  if (active) {
    // ---- layer 2: [16 x 64] x [64 x 32] ----
    v8f acc2[2] = {zero8(), zero8()};
    const _Float16* hr = hW + (size_t)lm * 64;
#pragma unroll
    for (int kt = 0; kt < 2; ++kt) {
      const int kb = kt * 32 + hi * 8;
      v16h a = mk16(load8h(hr + kb), load8h(hr + kb + 16));
#pragma unroll
      for (int nt = 0; nt < 2; ++nt) {
        const _Float16* wr = sW2 + (size_t)(nt * 16 + lm) * 64 + kb;
        v16h b = mk16(load8h(wr), load8h(wr + 16));
        acc2[nt] = wmma16x16x32(a, b, acc2[nt]);
      }
    }
    // bias + ReLU + LayerNorm(32)
    const float b20 = A.b2[lm], b21 = A.b2[16 + lm];
    const float g0 = A.g[lm], g1 = A.g[16 + lm];
    const float e0 = A.be[lm], e1 = A.be[16 + lm];
#pragma unroll
    for (int r = 0; r < 8; ++r) {
      float x0 = fmaxf(acc2[0][r] + b20, 0.f);
      float x1 = fmaxf(acc2[1][r] + b21, 0.f);
      float s = x0 + x1, q = x0 * x0 + x1 * x1;
#pragma unroll
      for (int msk = 1; msk < 16; msk <<= 1) {   // reduce over the 16-lane half
        s += __shfl_xor(s, msk, 32);
        q += __shfl_xor(q, msk, 32);
      }
      const float mean = s * (1.f / 32.f);
      const float var  = q * (1.f / 32.f) - mean * mean;
      const float inv  = rsqrtf(var + 1e-5f);
      const float y0 = g0 * (x0 - mean) * inv + e0;
      const float y1 = g1 * (x1 - mean) * inv + e1;
      if constexpr (!FUSE3) {
        float* orow = A.out + (size_t)(m0 + hi * 8 + r) * 32;
        orow[lm] = y0;
        orow[16 + lm] = y1;
      } else {
        hW[(size_t)(hi * 8 + r) * 32 + lm] = (_Float16)y0;
        hW[(size_t)(hi * 8 + r) * 32 + 16 + lm] = (_Float16)y1;
      }
    }
  }

  if constexpr (FUSE3) {
    __syncthreads();
    if (active) {
      // ---- fused output linear: [16 x 32] x [32 x 32] + b3 ----
      v8f acc3[2] = {zero8(), zero8()};
      const int kb = hi * 8;
      const _Float16* hr = hW + (size_t)lm * 32;
      v16h a = mk16(load8h(hr + kb), load8h(hr + kb + 16));
#pragma unroll
      for (int nt = 0; nt < 2; ++nt) {
        const _Float16* wr = sW3 + (size_t)(nt * 16 + lm) * 32 + kb;
        v16h b = mk16(load8h(wr), load8h(wr + 16));
        acc3[nt] = wmma16x16x32(a, b, acc3[nt]);
      }
#pragma unroll
      for (int nt = 0; nt < 2; ++nt) {
        const float bb = A.b3[nt * 16 + lm];
#pragma unroll
        for (int r = 0; r < 8; ++r)
          A.out[(size_t)(m0 + hi * 8 + r) * 32 + nt * 16 + lm] = acc3[nt][r] + bb;
      }
    }
  }
}

// ----------------------------- utility kernels -----------------------------

__global__ void zero_kernel(float* p, int n) {
  int i = blockIdx.x * 256 + threadIdx.x;
  if (i < n) p[i] = 0.f;
}

__global__ void add2_kernel(const float* a, const float* b, float* o, int n) {
  int i = blockIdx.x * 256 + threadIdx.x;
  if (i < n) o[i] = a[i] + b[i];
}

// Wt[n*K + k] = (f16) W[k*N + n]
__global__ void tcvt_kernel(const float* W, _Float16* Wt, int K, int N) {
  int i = blockIdx.x * 256 + threadIdx.x;
  if (i < K * N) {
    int k = i / N, n = i - k * N;
    Wt[(size_t)n * K + k] = (_Float16)W[i];
  }
}

// out64[r*64 + c] = f16( c<32 ? a[r*32+c] : b[r*32+c-32] )
__global__ void cat64_kernel(const float* a, const float* b, _Float16* o, int n64) {
  int i = blockIdx.x * 256 + threadIdx.x;
  if (i < n64) {
    int r = i >> 6, c = i & 63;
    float v = (c < 32) ? a[(size_t)r * 32 + c] : b[(size_t)r * 32 + (c - 32)];
    o[i] = (_Float16)v;
  }
}

__global__ void scatter32_kernel(const float* src, const int* idx, float* dst, int rows) {
  int i = blockIdx.x * 256 + threadIdx.x;
  if (i < rows * 32) {
    int r = i >> 5, c = i & 31;
    atomicAdd(dst + (size_t)idx[r] * 32 + c, src[i]);
  }
}

__global__ void countf_kernel(const int* idx, float* cnt, int n) {
  int i = blockIdx.x * 256 + threadIdx.x;
  if (i < n) atomicAdd(cnt + idx[i], 1.f);
}

// ------------------------------- host driver -------------------------------

static inline int gsz(int n) { return (n + 255) / 256; }

extern "C" void kernel_launch(void* const* d_in, const int* in_sizes, int n_in,
                              void* d_out, int out_size, void* d_ws, size_t ws_size,
                              hipStream_t stream) {
  (void)in_sizes; (void)n_in; (void)out_size; (void)ws_size;

  // ---- inputs (top-level dict insertion order) ----
  const float* x          = (const float*)d_in[0];
  const int*   edge_index = (const int*)  d_in[1];
  const float* edge_attr  = (const float*)d_in[2];
  const float* u_in       = (const float*)d_in[3];
  const float* embed      = (const float*)d_in[4];
  const int*   v_indices  = (const int*)  d_in[5];
  const int*   e_indices  = (const int*)  d_in[6];
  const int*   row = edge_index;            // edge_index[0]
  const int*   col = edge_index + kNE;      // edge_index[1]

  // params leaves (JAX pytree order: dicts sorted alphabetically)
  // core: edge, glob, node | dec: edge, glob, node | enc: edge, glob, node
  // then out_e, out_u, out_v.  Each MLP: W0,b0,W1,b1, ln_g, ln_b.
  const int P = 7;
  auto F = [&](int i) { return (const float*)d_in[P + i]; };
  // core.edge 0..5, core.glob 6..11, core.node 12..17,
  // dec.edge 18..23, dec.glob 24..29, dec.node 30..35,
  // enc.edge 36..41, enc.glob 42..47, enc.node 48..53,
  // out_e W/b 54..55, out_u 56..57, out_v 58..59

  // ---- workspace carve ----
  char* wsb = (char*)d_ws;
  size_t off = 0;
  auto take = [&](size_t bytes) -> void* {
    off = (off + 255) & ~(size_t)255;
    void* p = wsb + off;
    off += bytes;
    return p;
  };
  float* xpe    = (float*)take((size_t)kNV * 32 * 4);
  float* enc_x  = (float*)take((size_t)kNV * 32 * 4);
  float* enc_e  = (float*)take((size_t)kNE * 32 * 4);
  float* enc_u  = (float*)take((size_t)kNU * 32 * 4);
  float* latA_v = (float*)take((size_t)kNV * 32 * 4);
  float* latA_e = (float*)take((size_t)kNE * 32 * 4);
  float* latA_u = (float*)take((size_t)kNU * 32 * 4);
  float* latB_v = (float*)take((size_t)kNV * 32 * 4);
  float* latB_e = (float*)take((size_t)kNE * 32 * 4);
  float* latB_u = (float*)take((size_t)kNU * 32 * 4);
  float* agg    = (float*)take((size_t)kNV * 32 * 4);
  float* vsum   = (float*)take((size_t)kNU * 32 * 4);
  float* esum   = (float*)take((size_t)kNU * 32 * 4);
  float* vcnt   = (float*)take((size_t)kNU * 4);
  float* ecnt   = (float*)take((size_t)kNU * 4);
  _Float16* xv64 = (_Float16*)take((size_t)kNV * 64 * 2);
  _Float16* ea64 = (_Float16*)take((size_t)kNE * 64 * 2);
  _Float16* uu64 = (_Float16*)take((size_t)kNU * 64 * 2);
  _Float16* wt_ce1 = (_Float16*)take(64 * 256 * 2);
  _Float16* wt_ce2 = (_Float16*)take(32 * 64 * 2);
  _Float16* wt_cg1 = (_Float16*)take(64 * 128 * 2);
  _Float16* wt_cg2 = (_Float16*)take(32 * 64 * 2);
  _Float16* wt_cn1 = (_Float16*)take(64 * 160 * 2);
  _Float16* wt_cn2 = (_Float16*)take(32 * 64 * 2);
  _Float16* wt_de1 = (_Float16*)take(64 * 32 * 2);
  _Float16* wt_de2 = (_Float16*)take(32 * 64 * 2);
  _Float16* wt_dg1 = (_Float16*)take(64 * 32 * 2);
  _Float16* wt_dg2 = (_Float16*)take(32 * 64 * 2);
  _Float16* wt_dn1 = (_Float16*)take(64 * 32 * 2);
  _Float16* wt_dn2 = (_Float16*)take(32 * 64 * 2);
  _Float16* wt_ee1 = (_Float16*)take(64 * 32 * 2);
  _Float16* wt_ee2 = (_Float16*)take(32 * 64 * 2);
  _Float16* wt_eg1 = (_Float16*)take(64 * 32 * 2);
  _Float16* wt_eg2 = (_Float16*)take(32 * 64 * 2);
  _Float16* wt_en1 = (_Float16*)take(64 * 32 * 2);
  _Float16* wt_en2 = (_Float16*)take(32 * 64 * 2);
  _Float16* wt_oe  = (_Float16*)take(32 * 32 * 2);
  _Float16* wt_ou  = (_Float16*)take(32 * 32 * 2);
  _Float16* wt_ov  = (_Float16*)take(32 * 32 * 2);

  // ---- weight transpose + f16 convert ----
  auto tc = [&](const float* W, _Float16* Wt, int K, int N) {
    tcvt_kernel<<<gsz(K * N), 256, 0, stream>>>(W, Wt, K, N);
  };
  tc(F(0), wt_ce1, 256, 64);  tc(F(2), wt_ce2, 64, 32);
  tc(F(6), wt_cg1, 128, 64);  tc(F(8), wt_cg2, 64, 32);
  tc(F(12), wt_cn1, 160, 64); tc(F(14), wt_cn2, 64, 32);
  tc(F(18), wt_de1, 32, 64);  tc(F(20), wt_de2, 64, 32);
  tc(F(24), wt_dg1, 32, 64);  tc(F(26), wt_dg2, 64, 32);
  tc(F(30), wt_dn1, 32, 64);  tc(F(32), wt_dn2, 64, 32);
  tc(F(36), wt_ee1, 32, 64);  tc(F(38), wt_ee2, 64, 32);
  tc(F(42), wt_eg1, 32, 64);  tc(F(44), wt_eg2, 64, 32);
  tc(F(48), wt_en1, 32, 64);  tc(F(50), wt_en2, 64, 32);
  tc(F(54), wt_oe, 32, 32);   tc(F(56), wt_ou, 32, 32);   tc(F(58), wt_ov, 32, 32);

  add2_kernel<<<gsz(kNV * 32), 256, 0, stream>>>(x, embed, xpe, kNV * 32);

  MlpArgs base = {};
  base.xv64 = xv64; base.ea64 = ea64; base.uu64 = uu64;
  base.row = row; base.col = col; base.eidx = e_indices; base.vidx = v_indices;
  base.agg = agg; base.vsum = vsum; base.vcnt = vcnt; base.esum = esum; base.ecnt = ecnt;

  auto set_w = [&](MlpArgs& a, const _Float16* w1, int pb, const _Float16* w2) {
    a.W1t = w1; a.b1 = F(pb + 1); a.W2t = w2; a.b2 = F(pb + 3);
    a.g = F(pb + 4); a.be = F(pb + 5);
  };

  // ---- encoders ----
  {
    MlpArgs a = base; a.src = xpe; set_w(a, wt_en1, 48, wt_en2); a.out = enc_x; a.M = kNV;
    mlp_kernel<32, 0, false><<<gsz(kNV / 16 * 32), 256, 0, stream>>>(a);  // ceil(M/128) blocks
  }
  {
    MlpArgs a = base; a.src = edge_attr; set_w(a, wt_ee1, 36, wt_ee2); a.out = enc_e; a.M = kNE;
    mlp_kernel<32, 0, false><<<(kNE + 127) / 128, 256, 0, stream>>>(a);
  }
  {
    MlpArgs a = base; a.src = u_in; set_w(a, wt_eg1, 42, wt_eg2); a.out = enc_u; a.M = kNU;
    mlp_kernel<32, 0, false><<<1, 256, 0, stream>>>(a);
  }

  // ---- init latents + segment counts ----
  zero_kernel<<<gsz(kNV * 32), 256, 0, stream>>>(latA_v, kNV * 32);
  zero_kernel<<<gsz(kNE * 32), 256, 0, stream>>>(latA_e, kNE * 32);
  zero_kernel<<<gsz(kNU * 32), 256, 0, stream>>>(latA_u, kNU * 32);
  zero_kernel<<<1, 256, 0, stream>>>(vcnt, kNU);
  zero_kernel<<<1, 256, 0, stream>>>(ecnt, kNU);
  countf_kernel<<<gsz(kNV), 256, 0, stream>>>(v_indices, vcnt, kNV);
  countf_kernel<<<gsz(kNE), 256, 0, stream>>>(e_indices, ecnt, kNE);

  float* curv = latA_v; float* cure = latA_e; float* curu = latA_u;
  float* nxtv = latB_v; float* nxte = latB_e; float* nxtu = latB_u;

  // ---- core: 2 message-passing steps ----
  for (int step = 0; step < 2; ++step) {
    cat64_kernel<<<gsz(kNV * 64), 256, 0, stream>>>(enc_x, curv, xv64, kNV * 64);
    cat64_kernel<<<gsz(kNE * 64), 256, 0, stream>>>(enc_e, cure, ea64, kNE * 64);
    cat64_kernel<<<gsz(kNU * 64), 256, 0, stream>>>(enc_u, curu, uu64, kNU * 64);

    {  // edge model: K1 = 256 (xv[row]|xv[col]|ea|uu[eidx])
      MlpArgs a = base; set_w(a, wt_ce1, 0, wt_ce2); a.out = nxte; a.M = kNE;
      mlp_kernel<256, 1, false><<<(kNE + 127) / 128, 256, 0, stream>>>(a);
    }
    zero_kernel<<<gsz(kNV * 32), 256, 0, stream>>>(agg, kNV * 32);
    scatter32_kernel<<<gsz(kNE * 32), 256, 0, stream>>>(nxte, row, agg, kNE);

    {  // node model: K1 = 160 (xv|agg|uu[vidx])
      MlpArgs a = base; set_w(a, wt_cn1, 12, wt_cn2); a.out = nxtv; a.M = kNV;
      mlp_kernel<160, 2, false><<<(kNV + 127) / 128, 256, 0, stream>>>(a);
    }
    zero_kernel<<<gsz(kNU * 32), 256, 0, stream>>>(vsum, kNU * 32);
    zero_kernel<<<gsz(kNU * 32), 256, 0, stream>>>(esum, kNU * 32);
    scatter32_kernel<<<gsz(kNV * 32), 256, 0, stream>>>(nxtv, v_indices, vsum, kNV);
    scatter32_kernel<<<gsz(kNE * 32), 256, 0, stream>>>(nxte, e_indices, esum, kNE);

    {  // global model: K1 = 128 (uu|vmean|emean)
      MlpArgs a = base; set_w(a, wt_cg1, 6, wt_cg2); a.out = nxtu; a.M = kNU;
      mlp_kernel<128, 3, false><<<1, 256, 0, stream>>>(a);
    }
    float* t;
    t = curv; curv = nxtv; nxtv = t;
    t = cure; cure = nxte; nxte = t;
    t = curu; curu = nxtu; nxtu = t;
  }

  // ---- decoders + fused output linears ----
  float* out_v = (float*)d_out;
  float* out_e = out_v + (size_t)kNV * 32;
  float* out_u = out_e + (size_t)kNE * 32;
  {
    MlpArgs a = base; a.src = curv; set_w(a, wt_dn1, 30, wt_dn2);
    a.W3t = wt_ov; a.b3 = F(59); a.out = out_v; a.M = kNV;
    mlp_kernel<32, 0, true><<<(kNV + 127) / 128, 256, 0, stream>>>(a);
  }
  {
    MlpArgs a = base; a.src = cure; set_w(a, wt_de1, 18, wt_de2);
    a.W3t = wt_oe; a.b3 = F(55); a.out = out_e; a.M = kNE;
    mlp_kernel<32, 0, true><<<(kNE + 127) / 128, 256, 0, stream>>>(a);
  }
  {
    MlpArgs a = base; a.src = curu; set_w(a, wt_dg1, 24, wt_dg2);
    a.W3t = wt_ou; a.b3 = F(57); a.out = out_u; a.M = kNU;
    mlp_kernel<32, 0, true><<<1, 256, 0, stream>>>(a);
  }
}
